// GroupedQueryAttentionSolution_43215960932836
// MI455X (gfx1250) — matrix operand order repaired and checked by reading, not compiled, not run on hardware
//
#include <hip/hip_runtime.h>
#include <hip/hip_bf16.h>
#include <cstdint>

// ---------------------------------------------------------------------------
// GQA block for MI455X (gfx1250, wave32, WMMA, async-to-LDS copies).
// All matmuls run through v_wmma_f32_16x16x32_bf16 (fp32 accumulate).
// Tile copies use GLOBAL_LOAD_ASYNC_TO_LDS_B128 + s_wait_asynccnt (CDNA5 TDM-
// adjacent async path), double-buffered in the GEMM K-loop.
// ---------------------------------------------------------------------------

typedef __attribute__((ext_vector_type(16))) __bf16 v16bf;
typedef __attribute__((ext_vector_type(8)))  __bf16 v8bf;
typedef __attribute__((ext_vector_type(8)))  float  v8f;

constexpr int D_MODEL = 2048;
constexpr int N_HEADS = 32;
constexpr int N_KV    = 8;
constexpr int D_K     = 64;
constexpr int N_REP   = 4;
constexpr int BSZ     = 2;
constexpr int TSEQ    = 2048;
constexpr int MROWS   = BSZ * TSEQ;      // 4096
constexpr int KVDIM   = N_KV * D_K;      // 512

#define NEG_INF (-__builtin_inff())

// Async 16B global->LDS copy. LDS address = low 32 bits of flat shared ptr
// (ISA: flat LDS aperture keeps the byte offset in addr[31:0]).
#define ASYNC_LDS_B128(ldsOff, gPtr)                                        \
  asm volatile("global_load_async_to_lds_b128 %0, %1, off"                  \
               :: "v"(ldsOff), "v"(gPtr) : "memory")
#define WAIT_ASYNC0() asm volatile("s_wait_asynccnt 0x0" ::: "memory")

__device__ __forceinline__ uint32_t lds_off(const void* p) {
  return (uint32_t)(uintptr_t)p;
}

// ---------------------------------------------------------------------------
// fp32 -> bf16 streaming conversion (4 elems / thread, vectorized)
// ---------------------------------------------------------------------------
__global__ __launch_bounds__(256)
void gqa_cvt_bf16(const float* __restrict__ in, __bf16* __restrict__ out, int n) {
  int i = (blockIdx.x * blockDim.x + threadIdx.x) * 4;
  if (i < n) {
    float4 v = *reinterpret_cast<const float4*>(in + i);
    __bf16 t[4] = {(__bf16)v.x, (__bf16)v.y, (__bf16)v.z, (__bf16)v.w};
    *reinterpret_cast<float2*>(out + i) = *reinterpret_cast<const float2*>(t);
  }
}

// ---------------------------------------------------------------------------
// WMMA GEMM: C[M,N] = A[M,K] * W[N,K]^T   (torch Linear convention)
// 256 threads = 8 waves (4x2); block tile 128x64; K-step 64; wave tile 32x32.
// Double-buffered LDS fed by async-to-LDS copies: tile t+1 streams in while
// WMMAs consume tile t.
// ---------------------------------------------------------------------------
template<bool OUT_F32>
__global__ __launch_bounds__(256)
void gqa_gemm_wmma(const __bf16* __restrict__ A, const __bf16* __restrict__ W,
                   void* __restrict__ Cout, int Ndim, int Kdim) {
  constexpr int BM = 128, BN = 64, BK = 64;
  __shared__ __bf16 As[2][BM * BK];   // 16 KB per buffer
  __shared__ __bf16 Ws[2][BN * BK];   //  8 KB per buffer

  const int tid  = threadIdx.x;
  const int lane = tid & 31;
  const int wid  = tid >> 5;           // 0..7
  const int half = (lane >> 4) & 1;
  const int l16  = lane & 15;
  const int wm   = wid >> 1;           // 0..3  (wave row)
  const int wn   = wid & 1;            // 0..1  (wave col)
  const int n0   = blockIdx.x * BN;
  const int m0   = blockIdx.y * BM;

  v8f acc[2][2] = {};

  // cooperative async copy mapping
  const int arow = tid >> 1,  acol = (tid & 1) * 32;  // 128 x 64, 4 chunks/thr
  const int wrow = tid >> 2,  wcol = (tid & 3) * 16;  //  64 x 64, 2 chunks/thr

  auto load_tile = [&](int buf, int k0) {
    const __bf16* gA = A + (size_t)(m0 + arow) * Kdim + k0 + acol;
    const uint32_t la = lds_off(&As[buf][arow * BK + acol]);
#pragma unroll
    for (int c = 0; c < 4; ++c)
      ASYNC_LDS_B128(la + c * 16, gA + c * 8);
    const __bf16* gW = W + (size_t)(n0 + wrow) * Kdim + k0 + wcol;
    const uint32_t lw = lds_off(&Ws[buf][wrow * BK + wcol]);
    ASYNC_LDS_B128(lw, gW);
    ASYNC_LDS_B128(lw + 16, gW + 8);
  };

  load_tile(0, 0);
  WAIT_ASYNC0();
  __syncthreads();

  const int iters = Kdim / BK;
  for (int it = 0; it < iters; ++it) {
    const int buf = it & 1;
    if (it + 1 < iters) load_tile(buf ^ 1, (it + 1) * BK);  // overlap copy

#pragma unroll
    for (int ks = 0; ks < BK; ks += 32) {
      // A frags: 16-bit A 16x32 layout (lane-half splits K chunks)
      v16bf afrag[2];
#pragma unroll
      for (int mi = 0; mi < 2; ++mi) {
        const __bf16* p = &As[buf][(wm * 32 + mi * 16 + l16) * BK + ks + half * 8];
        v8bf lo = *reinterpret_cast<const v8bf*>(p);
        v8bf hi = *reinterpret_cast<const v8bf*>(p + 16);
#pragma unroll
        for (int e = 0; e < 8; ++e) { afrag[mi][e] = lo[e]; afrag[mi][e + 8] = hi[e]; }
      }
      // B frags: 32x16, lanes 0-15 K 0-7,16-23, lanes 16-31 K 8-15,24-31
      v16bf bfrag[2];
#pragma unroll
      for (int nj = 0; nj < 2; ++nj) {
        const __bf16* p = &Ws[buf][(wn * 32 + nj * 16 + l16) * BK + ks + half * 8];
        v8bf lo = *reinterpret_cast<const v8bf*>(p);
        v8bf hi = *reinterpret_cast<const v8bf*>(p + 16);
#pragma unroll
        for (int e = 0; e < 8; ++e) { bfrag[nj][e] = lo[e]; bfrag[nj][e + 8] = hi[e]; }
      }
#pragma unroll
      for (int mi = 0; mi < 2; ++mi)
#pragma unroll
        for (int nj = 0; nj < 2; ++nj)
          acc[mi][nj] = __builtin_amdgcn_wmma_f32_16x16x32_bf16(
              false, afrag[mi], false, bfrag[nj], (short)0, acc[mi][nj], false, false);
    }
    WAIT_ASYNC0();       // my copies into buf^1 done
    __syncthreads();     // everyone done reading buf / writing buf^1
  }

  // C layout: VGPR i -> row i (lanes 0-15) / i+8 (lanes 16-31), col = lane%16
#pragma unroll
  for (int mi = 0; mi < 2; ++mi)
#pragma unroll
    for (int nj = 0; nj < 2; ++nj)
#pragma unroll
      for (int i = 0; i < 8; ++i) {
        const int row = m0 + wm * 32 + mi * 16 + i + half * 8;
        const int col = n0 + wn * 32 + nj * 16 + l16;
        if (OUT_F32)
          ((float*)Cout)[(size_t)row * Ndim + col] = acc[mi][nj][i];
        else
          ((__bf16*)Cout)[(size_t)row * Ndim + col] = (__bf16)acc[mi][nj][i];
      }
}

// ---------------------------------------------------------------------------
// Flash attention (causal, grouped KV). Block = 128 threads (4 waves),
// one block per (batch, q-head, 64 query rows); each wave owns 16 rows.
// K tile staged via async-to-LDS; V staged transposed; softmax in exp2 domain.
// ---------------------------------------------------------------------------
__global__ __launch_bounds__(128)
void gqa_flash_wmma(const __bf16* __restrict__ Qb, const __bf16* __restrict__ Kb,
                    const __bf16* __restrict__ Vb, __bf16* __restrict__ Ob) {
  __shared__ __bf16 Kt[32 * 64];       // [key][dim] row-major
  __shared__ __bf16 Vt[64 * 32];       // [dim][key] transposed
  __shared__ __bf16 Pl[4 * 16 * 32];   // per-wave P staging (C->A relayout)

  const int tid  = threadIdx.x;
  const int lane = tid & 31;
  const int wid  = tid >> 5;
  const int half = (lane >> 4) & 1;
  const int l16  = lane & 15;
  const int q0   = blockIdx.x * 64;
  const int h    = blockIdx.y;
  const int b    = blockIdx.z;
  const int g    = h / N_REP;
  const float qk_scale = 0.125f * 1.44269504088896340736f;  // 1/sqrt(64)*log2(e)

  // Preload this wave's Q (16 rows x 64 dims) as two 16x32 A-fragments.
  v16bf qf[2];
  {
    const int qrow = q0 + wid * 16 + l16;
    const __bf16* qp = Qb + (size_t)(b * TSEQ + qrow) * D_MODEL + h * D_K;
#pragma unroll
    for (int ks = 0; ks < 2; ++ks) {
      v8bf lo = *reinterpret_cast<const v8bf*>(qp + ks * 32 + half * 8);
      v8bf hi = *reinterpret_cast<const v8bf*>(qp + ks * 32 + half * 8 + 16);
#pragma unroll
      for (int e = 0; e < 8; ++e) { qf[ks][e] = lo[e]; qf[ks][e + 8] = hi[e]; }
    }
  }

  v8f oacc[4] = {};
  float mrun[8], lrun[8];
#pragma unroll
  for (int i = 0; i < 8; ++i) { mrun[i] = NEG_INF; lrun[i] = 0.f; }

  const int kvkey = tid >> 2;          // 0..31
  const int kvdim = (tid & 3) * 16;    // 0,16,32,48

  for (int j0 = 0; j0 < q0 + 64; j0 += 32) {
    // K tile: async global->LDS (row-major, contiguous 16B chunks)
    {
      const __bf16* kp = Kb + (size_t)(b * TSEQ + j0 + kvkey) * KVDIM + g * D_K + kvdim;
      const uint32_t lk = lds_off(&Kt[kvkey * 64 + kvdim]);
      ASYNC_LDS_B128(lk, kp);
      ASYNC_LDS_B128(lk + 16, kp + 8);
    }
    // V tile: load to registers, scatter transposed into LDS
    {
      const __bf16* vp = Vb + (size_t)(b * TSEQ + j0 + kvkey) * KVDIM + g * D_K + kvdim;
      const float4* vg = reinterpret_cast<const float4*>(vp);
      float4 va = vg[0], vb2 = vg[1];
      __bf16 tmp[16];
      *reinterpret_cast<float4*>(tmp)     = va;
      *reinterpret_cast<float4*>(tmp + 8) = vb2;
#pragma unroll
      for (int e = 0; e < 16; ++e) Vt[(kvdim + e) * 32 + kvkey] = tmp[e];
    }
    WAIT_ASYNC0();
    __syncthreads();

    // S = Q K^T : two 16x16 fragments (32 keys), K-dim 64 = 2 wmma each
    v8f sf[2];
#pragma unroll
    for (int jf = 0; jf < 2; ++jf) {
      v8f s = {};
#pragma unroll
      for (int ks = 0; ks < 2; ++ks) {
        v16bf kfrag;
        const __bf16* p = &Kt[(jf * 16 + l16) * 64 + ks * 32 + half * 8];
        v8bf lo = *reinterpret_cast<const v8bf*>(p);
        v8bf hi = *reinterpret_cast<const v8bf*>(p + 16);
#pragma unroll
        for (int e = 0; e < 8; ++e) { kfrag[e] = lo[e]; kfrag[e + 8] = hi[e]; }
        s = __builtin_amdgcn_wmma_f32_16x16x32_bf16(
            false, qf[ks], false, kfrag, (short)0, s, false, false);
      }
      sf[jf] = s;
    }

    // causal mask + online softmax in exp2 domain
    float p0[8], p1[8];
#pragma unroll
    for (int i = 0; i < 8; ++i) {
      const int row = q0 + wid * 16 + i + half * 8;
      const int c0  = j0 + l16;
      const int c1  = j0 + 16 + l16;
      float s0 = (c0 <= row) ? sf[0][i] * qk_scale : NEG_INF;
      float s1 = (c1 <= row) ? sf[1][i] * qk_scale : NEG_INF;
      float vmax = fmaxf(s0, s1);
#pragma unroll
      for (int m = 1; m < 16; m <<= 1) vmax = fmaxf(vmax, __shfl_xor(vmax, m, 32));
      const float mnew = fmaxf(mrun[i], vmax);
      const float corr = (mnew == NEG_INF) ? 1.f : exp2f(mrun[i] - mnew);
      const float e0 = (s0 == NEG_INF) ? 0.f : exp2f(s0 - mnew);
      const float e1 = (s1 == NEG_INF) ? 0.f : exp2f(s1 - mnew);
      float ps = e0 + e1;
#pragma unroll
      for (int m = 1; m < 16; m <<= 1) ps += __shfl_xor(ps, m, 32);
      lrun[i] = lrun[i] * corr + ps;
      mrun[i] = mnew;
      p0[i] = e0; p1[i] = e1;
#pragma unroll
      for (int nf = 0; nf < 4; ++nf) oacc[nf][i] *= corr;
    }

    // P: C-layout -> A-layout via per-wave LDS region
#pragma unroll
    for (int i = 0; i < 8; ++i) {
      Pl[wid * 512 + (i + half * 8) * 32 + l16]      = (__bf16)p0[i];
      Pl[wid * 512 + (i + half * 8) * 32 + 16 + l16] = (__bf16)p1[i];
    }
    v16bf pa;
    {
      const __bf16* p = &Pl[wid * 512 + l16 * 32 + half * 8];
      v8bf lo = *reinterpret_cast<const v8bf*>(p);
      v8bf hi = *reinterpret_cast<const v8bf*>(p + 16);
#pragma unroll
      for (int e = 0; e < 8; ++e) { pa[e] = lo[e]; pa[e + 8] = hi[e]; }
    }

    // O += P @ V  (V transposed in LDS: B-frag chunks contiguous over keys)
#pragma unroll
    for (int nf = 0; nf < 4; ++nf) {
      v16bf vfrag;
      const __bf16* p = &Vt[(nf * 16 + l16) * 32 + half * 8];
      v8bf lo = *reinterpret_cast<const v8bf*>(p);
      v8bf hi = *reinterpret_cast<const v8bf*>(p + 16);
#pragma unroll
      for (int e = 0; e < 8; ++e) { vfrag[e] = lo[e]; vfrag[e + 8] = hi[e]; }
      oacc[nf] = __builtin_amdgcn_wmma_f32_16x16x32_bf16(
          false, pa, false, vfrag, (short)0, oacc[nf], false, false);
    }
    __syncthreads();
  }

  // normalize rows and store bf16 head output
#pragma unroll
  for (int i = 0; i < 8; ++i) {
    const float rinv = (lrun[i] > 0.f) ? 1.f / lrun[i] : 0.f;
    const int row = q0 + wid * 16 + i + half * 8;
#pragma unroll
    for (int nf = 0; nf < 4; ++nf)
      Ob[(size_t)(b * TSEQ + row) * D_MODEL + h * D_K + nf * 16 + l16] =
          (__bf16)(oacc[nf][i] * rinv);
  }
}

// ---------------------------------------------------------------------------
// Launcher
// ---------------------------------------------------------------------------
extern "C" void kernel_launch(void* const* d_in, const int* in_sizes, int n_in,
                              void* d_out, int out_size, void* d_ws, size_t ws_size,
                              hipStream_t stream) {
  (void)in_sizes; (void)n_in; (void)out_size; (void)ws_size;
  const float* x  = (const float*)d_in[0];
  const float* Wq = (const float*)d_in[1];
  const float* Wk = (const float*)d_in[2];
  const float* Wv = (const float*)d_in[3];
  const float* Wo = (const float*)d_in[4];

  char* ws = (char*)d_ws;
  auto alloc = [&](size_t elems) {
    __bf16* p = reinterpret_cast<__bf16*>(ws);
    ws += elems * 2;
    return p;
  };
  __bf16* xb  = alloc((size_t)MROWS * D_MODEL);
  __bf16* Wqb = alloc((size_t)D_MODEL * D_MODEL);
  __bf16* Wkb = alloc((size_t)KVDIM * D_MODEL);
  __bf16* Wvb = alloc((size_t)KVDIM * D_MODEL);
  __bf16* Wob = alloc((size_t)D_MODEL * D_MODEL);
  __bf16* Qb  = alloc((size_t)MROWS * D_MODEL);
  __bf16* Kb  = alloc((size_t)MROWS * KVDIM);
  __bf16* Vb  = alloc((size_t)MROWS * KVDIM);
  __bf16* Ob  = alloc((size_t)MROWS * D_MODEL);

  auto cvt = [&](const float* in, __bf16* out, size_t n) {
    gqa_cvt_bf16<<<dim3((unsigned)((n + 1023) / 1024)), dim3(256), 0, stream>>>(
        in, out, (int)n);
  };
  cvt(x,  xb,  (size_t)MROWS * D_MODEL);
  cvt(Wq, Wqb, (size_t)D_MODEL * D_MODEL);
  cvt(Wk, Wkb, (size_t)KVDIM * D_MODEL);
  cvt(Wv, Wvb, (size_t)KVDIM * D_MODEL);
  cvt(Wo, Wob, (size_t)D_MODEL * D_MODEL);

  gqa_gemm_wmma<false><<<dim3(D_MODEL / 64, MROWS / 128), 256, 0, stream>>>(
      xb, Wqb, Qb, D_MODEL, D_MODEL);
  gqa_gemm_wmma<false><<<dim3(KVDIM / 64, MROWS / 128), 256, 0, stream>>>(
      xb, Wkb, Kb, KVDIM, D_MODEL);
  gqa_gemm_wmma<false><<<dim3(KVDIM / 64, MROWS / 128), 256, 0, stream>>>(
      xb, Wvb, Vb, KVDIM, D_MODEL);

  gqa_flash_wmma<<<dim3(TSEQ / 64, N_HEADS, BSZ), 128, 0, stream>>>(Qb, Kb, Vb, Ob);

  gqa_gemm_wmma<true><<<dim3(D_MODEL / 64, MROWS / 128), 256, 0, stream>>>(
      Ob, Wob, d_out, D_MODEL, D_MODEL);
}
